// DgcnNetwork_54150947668229
// MI455X (gfx1250) — matrix operand order repaired and checked by reading, not compiled, not run on hardware
//
#include <hip/hip_runtime.h>
#include <hip/hip_bf16.h>

// ---------------------------------------------------------------------------
// DGCNN forward for MI455X (gfx1250, wave32, WMMA 16x16x32 f16 -> f32 accum)
//
// v3: + async LDS staging of weight tiles (global_load_async_to_lds_b128,
//       ASYNCcnt-tracked, overlapped with the A-tile gather)
//     + 8-channel-chunk vectorized gathers (b128 global loads, mergeable
//       contiguous ds stores into the fragment-major A layout)
//     + BN scale/shift staged in LDS for edge_gemm2
// ---------------------------------------------------------------------------

typedef __attribute__((ext_vector_type(16))) _Float16 v16h;
typedef __attribute__((ext_vector_type(8)))  float    v8f;

__device__ __forceinline__ v8f wmma_f16(v16h a, v16h b, v8f c) {
  return __builtin_amdgcn_wmma_f32_16x16x32_f16(false, a, false, b,
                                                (short)0, c, false, false);
}

// per-lane async copy: LDS[lds_off] = MEM[base + byte_off], 16 bytes
__device__ __forceinline__ void async_load_b128(unsigned lds_off,
                                                const void* base,
                                                unsigned byte_off) {
  asm volatile("global_load_async_to_lds_b128 %0, %1, %2"
               :: "v"(lds_off), "v"(byte_off), "s"(base)
               : "memory");
}
__device__ __forceinline__ void wait_async0() {
  asm volatile("s_wait_asynccnt 0x0" ::: "memory");
}
__device__ __forceinline__ unsigned lds_addr_of(const void* p) {
  return (unsigned)(uintptr_t)p;   // flat LDS aperture: low 32 bits = LDS addr
}

// monotonic float<->uint encoding for atomic max on floats
__device__ __forceinline__ unsigned encf(float x) {
  unsigned u = __float_as_uint(x);
  return (u & 0x80000000u) ? ~u : (u | 0x80000000u);
}
__device__ __forceinline__ float decf(unsigned e) {
  unsigned u = (e & 0x80000000u) ? (e ^ 0x80000000u) : ~e;
  return __uint_as_float(u);
}

// A-fragment scatter position for element (row m, depth k) of a 16xK tile.
// For an aligned 8-chunk (k0..k0+7), positions are contiguous (base + j).
__device__ __forceinline__ int afrag_pos(int m, int k) {
  int kb = k >> 5, kl = k & 31;
  int lane = m + (((kl >> 3) & 1) << 4);
  int t = ((kl >> 4) << 3) + (kl & 7);
  return kb * 512 + lane * 16 + t;
}

// ---------------------------------------------------------------------------
// utility kernels
// ---------------------------------------------------------------------------
__global__ void fill_u32(unsigned* __restrict__ p, unsigned v, long long n) {
  long long i = (long long)blockIdx.x * blockDim.x + threadIdx.x;
  if (i < n) p[i] = v;
}

// Swizzle a [K][Cout] f32 weight into fragment-major f16.
__global__ void weight_swizzle(const float* __restrict__ src,
                               _Float16* __restrict__ dst,
                               int Cout, long long total) {
  long long gid = (long long)blockIdx.x * blockDim.x + threadIdx.x;
  if (gid >= total) return;
  int t = (int)(gid & 15);
  int lane = (int)((gid >> 4) & 31);
  long long rest = gid >> 9;
  int ntiles = Cout >> 4;
  int ntile = (int)(rest % ntiles);
  long long kb = rest / ntiles;
  int k = (int)(kb * 32) + ((t >> 3) << 4) + ((lane >> 4) << 3) + (t & 7);
  int n = ntile * 16 + (lane & 15);
  dst[gid] = (_Float16)src[(long long)k * Cout + n];
}

// stats: [C] sums then [C] sumsqs -> bn: [C] scale then [C] shift
__global__ void bn_finalize(const float* __restrict__ stats,
                            const float* __restrict__ gamma,
                            const float* __restrict__ beta,
                            float count, int C, float* __restrict__ bn) {
  int c = threadIdx.x;
  if (c < C) {
    float mean = stats[c] / count;
    float var  = fmaxf(stats[C + c] / count - mean * mean, 0.f);
    float sc   = gamma[c] * rsqrtf(var + 1e-5f);
    bn[c]     = sc;
    bn[C + c] = beta[c] - mean * sc;
  }
}

// ---------------------------------------------------------------------------
// EdgeConv block 1: A = [x_i | x_j - x_i] (16 edges x 2*CIN), B = W (2*CIN x 64)
// one wave = one 16-edge tile; 8 waves / block
// ---------------------------------------------------------------------------
template <int CIN, typename XT>
__global__ __launch_bounds__(256)
void edge_gemm1(const XT* __restrict__ xf,
                const long long* __restrict__ src,
                const long long* __restrict__ dst,
                const _Float16* __restrict__ Wsw,    // swizzled [KB][4][32][16] f16
                const float* __restrict__ bias,      // [64]
                _Float16* __restrict__ hout,         // [E][64]
                float* __restrict__ stats,           // [64]+[64]
                long long E) {
  constexpr int K = 2 * CIN;
  constexpr int KB = K / 32;
  constexpr int CH = CIN / 8;                        // 8-channel chunks per row
  __shared__ alignas(32) _Float16 sA[8][16 * K];
  __shared__ alignas(32) _Float16 sW[K * 64];
  __shared__ float sSum[64], sSq[64];

  const int tid = threadIdx.x, wave = tid >> 5, lane = tid & 31;
  const int m = lane & 15, half = lane >> 4;

  for (int i = tid; i < 64; i += 256) { sSum[i] = 0.f; sSq[i] = 0.f; }

  // async-stage swizzled W into LDS (overlaps with gather below)
  for (int i = tid; i < K * 64 / 8; i += 256)
    async_load_b128(lds_addr_of(&sW[i * 8]), Wsw, (unsigned)(i * 16));

  const long long tile = (long long)blockIdx.x * 8 + wave;
  const long long e0 = tile * 16;

  // gather A into LDS in fragment order, 8 channels per step (zero-pad past E)
  for (int i = lane; i < 16 * CH; i += 32) {
    int er = i / CH, c0 = (i - er * CH) * 8;
    long long e = e0 + er;
    float xd[8], xs[8];
    if (e < E) {
      long long dn = dst[e], sn = src[e];
      if constexpr (sizeof(XT) == 2) {
        union { uint4 u; _Float16 h[8]; } bd, bs;
        bd.u = *(const uint4*)(xf + dn * CIN + c0);
        bs.u = *(const uint4*)(xf + sn * CIN + c0);
#pragma unroll
        for (int j = 0; j < 8; j++) { xd[j] = (float)bd.h[j]; xs[j] = (float)bs.h[j]; }
      } else {
        const float4* pd = (const float4*)(xf + dn * CIN + c0);
        const float4* ps = (const float4*)(xf + sn * CIN + c0);
        float4 d0 = pd[0], d1 = pd[1], s0 = ps[0], s1 = ps[1];
        xd[0] = d0.x; xd[1] = d0.y; xd[2] = d0.z; xd[3] = d0.w;
        xd[4] = d1.x; xd[5] = d1.y; xd[6] = d1.z; xd[7] = d1.w;
        xs[0] = s0.x; xs[1] = s0.y; xs[2] = s0.z; xs[3] = s0.w;
        xs[4] = s1.x; xs[5] = s1.y; xs[6] = s1.z; xs[7] = s1.w;
      }
    } else {
#pragma unroll
      for (int j = 0; j < 8; j++) { xd[j] = 0.f; xs[j] = 0.f; }
    }
    const int pa = afrag_pos(er, c0);        // contiguous for the 8-chunk
    const int pb = afrag_pos(er, CIN + c0);
#pragma unroll
    for (int j = 0; j < 8; j++) {
      sA[wave][pa + j] = (_Float16)xd[j];
      sA[wave][pb + j] = (_Float16)(xs[j] - xd[j]);
    }
  }
  wait_async0();
  __syncthreads();

  v16h areg[KB];
#pragma unroll
  for (int kb = 0; kb < KB; kb++)
    areg[kb] = *reinterpret_cast<const v16h*>(&sA[wave][kb * 512 + lane * 16]);

  float bch[4];
#pragma unroll
  for (int nt = 0; nt < 4; nt++) bch[nt] = bias[nt * 16 + m];

  v8f acc[4];
#pragma unroll
  for (int nt = 0; nt < 4; nt++) {
    v8f c = {};
#pragma unroll
    for (int kb = 0; kb < KB; kb++) {
      v16h b = *reinterpret_cast<const v16h*>(&sW[(kb * 4 + nt) * 512 + lane * 16]);
      c = wmma_f16(areg[kb], b, c);
    }
    acc[nt] = c;
  }

  const bool full = (e0 + 16 <= E);
#pragma unroll
  for (int nt = 0; nt < 4; nt++) {
    const int ch = nt * 16 + m;
    float s = 0.f, q = 0.f;
    if (full) {
#pragma unroll
      for (int r = 0; r < 8; r++) {
        long long e = e0 + r + half * 8;
        float v = fmaxf(acc[nt][r] + bch[nt], 0.f);
        hout[e * 64 + ch] = (_Float16)v;
        s += v; q += v * v;
      }
    } else {
#pragma unroll
      for (int r = 0; r < 8; r++) {
        long long e = e0 + r + half * 8;
        float v = fmaxf(acc[nt][r] + bch[nt], 0.f);
        if (e < E) { hout[e * 64 + ch] = (_Float16)v; s += v; q += v * v; }
      }
    }
    s += __shfl_xor(s, 16);
    q += __shfl_xor(q, 16);
    if (half == 0) { atomicAdd(&sSum[ch], s); atomicAdd(&sSq[ch], q); }
  }
  __syncthreads();
  for (int i = tid; i < 64; i += 256) {
    atomicAdd(&stats[i],      sSum[i]);
    atomicAdd(&stats[64 + i], sSq[i]);
  }
}

// ---------------------------------------------------------------------------
// EdgeConv block 2: apply BN1 (staged in LDS) while building fragment-major A,
// GEMM with W2 (64x64), ReLU, write back IN PLACE, accumulate BN2 stats.
// ---------------------------------------------------------------------------
__global__ __launch_bounds__(256)
void edge_gemm2(_Float16* __restrict__ hedge,
                const float* __restrict__ bn,     // scale[64], shift[64]
                const _Float16* __restrict__ Wsw, // swizzled 64x64
                const float* __restrict__ bias,
                float* __restrict__ stats,
                long long E) {
  constexpr int K = 64, KB = 2;
  __shared__ alignas(32) _Float16 sA[8][16 * K];
  __shared__ alignas(32) _Float16 sW[K * 64];
  __shared__ float sSum[64], sSq[64];
  __shared__ float sBN[128];

  const int tid = threadIdx.x, wave = tid >> 5, lane = tid & 31;
  const int m = lane & 15, half = lane >> 4;

  for (int i = tid; i < 64; i += 256) { sSum[i] = 0.f; sSq[i] = 0.f; }
  if (tid < 128) sBN[tid] = bn[tid];

  // async-stage swizzled W into LDS
  for (int i = tid; i < K * 64 / 8; i += 256)
    async_load_b128(lds_addr_of(&sW[i * 8]), Wsw, (unsigned)(i * 16));
  wait_async0();
  __syncthreads();   // sBN + sW ready

  const long long tile = (long long)blockIdx.x * 8 + wave;
  const long long e0 = tile * 16;

  for (int i = lane; i < 16 * 8; i += 32) {   // 16 rows x 8 chunks of 8
    int er = i >> 3, c0 = (i & 7) * 8;
    long long e = e0 + er;
    float v[8];
    if (e < E) {
      union { uint4 u; _Float16 h[8]; } blk;
      blk.u = *(const uint4*)(hedge + e * 64 + c0);
#pragma unroll
      for (int j = 0; j < 8; j++)
        v[j] = (float)blk.h[j] * sBN[c0 + j] + sBN[64 + c0 + j];
    } else {
#pragma unroll
      for (int j = 0; j < 8; j++) v[j] = 0.f;
    }
    const int pa = afrag_pos(er, c0);
#pragma unroll
    for (int j = 0; j < 8; j++) sA[wave][pa + j] = (_Float16)v[j];
  }
  __syncthreads();

  v16h areg[KB];
#pragma unroll
  for (int kb = 0; kb < KB; kb++)
    areg[kb] = *reinterpret_cast<const v16h*>(&sA[wave][kb * 512 + lane * 16]);

  float bch[4];
#pragma unroll
  for (int nt = 0; nt < 4; nt++) bch[nt] = bias[nt * 16 + m];

  v8f acc[4];
#pragma unroll
  for (int nt = 0; nt < 4; nt++) {
    v8f c = {};
#pragma unroll
    for (int kb = 0; kb < KB; kb++) {
      v16h b = *reinterpret_cast<const v16h*>(&sW[(kb * 4 + nt) * 512 + lane * 16]);
      c = wmma_f16(areg[kb], b, c);
    }
    acc[nt] = c;
  }

  const bool full = (e0 + 16 <= E);
#pragma unroll
  for (int nt = 0; nt < 4; nt++) {
    const int ch = nt * 16 + m;
    float s = 0.f, q = 0.f;
    if (full) {
#pragma unroll
      for (int r = 0; r < 8; r++) {
        long long e = e0 + r + half * 8;
        float v = fmaxf(acc[nt][r] + bch[nt], 0.f);
        hedge[e * 64 + ch] = (_Float16)v;   // in-place: tile rows are private
        s += v; q += v * v;
      }
    } else {
#pragma unroll
      for (int r = 0; r < 8; r++) {
        long long e = e0 + r + half * 8;
        float v = fmaxf(acc[nt][r] + bch[nt], 0.f);
        if (e < E) { hedge[e * 64 + ch] = (_Float16)v; s += v; q += v * v; }
      }
    }
    s += __shfl_xor(s, 16);
    q += __shfl_xor(q, 16);
    if (half == 0) { atomicAdd(&sSum[ch], s); atomicAdd(&sSq[ch], q); }
  }
  __syncthreads();
  for (int i = tid; i < 64; i += 256) {
    atomicAdd(&stats[i],      sSum[i]);
    atomicAdd(&stats[64 + i], sSq[i]);
  }
}

// apply BN2 and scatter-max into encoded node buffer (4 channels / thread)
__global__ void scatter_max(const _Float16* __restrict__ hedge,
                            const float* __restrict__ bn,
                            const long long* __restrict__ dst,
                            unsigned* __restrict__ nodeEnc,
                            long long total /* E*16 */) {
  long long gid = (long long)blockIdx.x * blockDim.x + threadIdx.x;
  if (gid >= total) return;
  long long e = gid >> 4;
  int c0 = (int)(gid & 15) << 2;
  const _Float16* hp = hedge + e * 64 + c0;
  long long d = dst[e];
  unsigned* np = nodeEnc + d * 64 + c0;
#pragma unroll
  for (int j = 0; j < 4; j++) {
    float v = (float)hp[j] * bn[c0 + j] + bn[64 + c0 + j];
    atomicMax(&np[j], encf(v));
  }
}

// decode node buffer; isolated nodes (-inf) -> 0; pack pairs into b32 stores
__global__ void decode_nodes(const unsigned* __restrict__ enc,
                             unsigned* __restrict__ out, long long pairs) {
  long long gid = (long long)blockIdx.x * blockDim.x + threadIdx.x;
  if (gid >= pairs) return;
  float v0 = decf(enc[gid * 2]);
  float v1 = decf(enc[gid * 2 + 1]);
  if (!isfinite(v0)) v0 = 0.f;
  if (!isfinite(v1)) v1 = 0.f;
  union { _Float16 h[2]; unsigned u; } pk;
  pk.h[0] = (_Float16)v0;
  pk.h[1] = (_Float16)v1;
  out[gid] = pk.u;
}

// ---------------------------------------------------------------------------
// lin1: A = cat[x1,x2,x3] (16 nodes x 192), B = W (192x256) fragment-major in
// global (L2-resident). block = 128 threads = 4 waves.
// ---------------------------------------------------------------------------
__global__ __launch_bounds__(128)
void lin1_gemm(const _Float16* __restrict__ x1, const _Float16* __restrict__ x2,
               const _Float16* __restrict__ x3,
               const _Float16* __restrict__ Wsw,   // [6][16][32][16] f16
               const float* __restrict__ bias,     // [256]
               _Float16* __restrict__ hout,        // [N][256]
               float* __restrict__ stats,          // [256]+[256]
               long long N) {
  constexpr int KB = 6;
  __shared__ alignas(32) _Float16 sA[16 * 192];
  __shared__ float sSum[256], sSq[256];

  const int tid = threadIdx.x, wave = tid >> 5, lane = tid & 31;
  const int m = lane & 15, half = lane >> 4;
  for (int i = tid; i < 256; i += 128) { sSum[i] = 0.f; sSq[i] = 0.f; }

  const long long n0 = (long long)blockIdx.x * 16;
  for (int i = tid; i < 16 * 24; i += 128) {     // 16 rows x 24 chunks of 8
    int r = i / 24, c = i - r * 24;
    int k0 = c * 8;
    long long n = n0 + r;
    float v[8];
    if (n < N) {
      const _Float16* sp = (k0 < 64) ? (x1 + n * 64 + k0)
                         : (k0 < 128) ? (x2 + n * 64 + (k0 - 64))
                                      : (x3 + n * 64 + (k0 - 128));
      union { uint4 u; _Float16 h[8]; } blk;
      blk.u = *(const uint4*)sp;
#pragma unroll
      for (int j = 0; j < 8; j++) v[j] = (float)blk.h[j];
    } else {
#pragma unroll
      for (int j = 0; j < 8; j++) v[j] = 0.f;
    }
    const int pa = afrag_pos(r, k0);
#pragma unroll
    for (int j = 0; j < 8; j++) sA[pa + j] = (_Float16)v[j];
  }
  __syncthreads();

  v16h areg[KB];
#pragma unroll
  for (int kb = 0; kb < KB; kb++)
    areg[kb] = *reinterpret_cast<const v16h*>(&sA[kb * 512 + lane * 16]);

  const int nbase = wave * 64;
  float bch[4];
#pragma unroll
  for (int nt = 0; nt < 4; nt++) bch[nt] = bias[nbase + nt * 16 + m];

  const bool full = (n0 + 16 <= N);
#pragma unroll
  for (int nt = 0; nt < 4; nt++) {
    const int ntile = (nbase >> 4) + nt;
    v8f acc = {};
#pragma unroll
    for (int kb = 0; kb < KB; kb++) {
      v16h b = *reinterpret_cast<const v16h*>(
          &Wsw[((long long)kb * 16 + ntile) * 512 + lane * 16]);
      acc = wmma_f16(areg[kb], b, acc);
    }
    const int ch = nbase + nt * 16 + m;
    float s = 0.f, q = 0.f;
    if (full) {
#pragma unroll
      for (int r = 0; r < 8; r++) {
        long long n = n0 + r + half * 8;
        float v = fmaxf(acc[r] + bch[nt], 0.f);
        hout[n * 256 + ch] = (_Float16)v;
        s += v; q += v * v;
      }
    } else {
#pragma unroll
      for (int r = 0; r < 8; r++) {
        long long n = n0 + r + half * 8;
        float v = fmaxf(acc[r] + bch[nt], 0.f);
        if (n < N) { hout[n * 256 + ch] = (_Float16)v; s += v; q += v * v; }
      }
    }
    s += __shfl_xor(s, 16);
    q += __shfl_xor(q, 16);
    if (half == 0) { atomicAdd(&sSum[ch], s); atomicAdd(&sSq[ch], q); }
  }
  __syncthreads();
  for (int i = tid; i < 256; i += 128) {
    atomicAdd(&stats[i],       sSum[i]);
    atomicAdd(&stats[256 + i], sSq[i]);
  }
}

// apply lin1 BN, accumulate per-graph sums + counts (mean pool)
__global__ void pool_kernel(const _Float16* __restrict__ hnode,
                            const float* __restrict__ bn,        // [256]+[256]
                            const long long* __restrict__ batch,
                            float* __restrict__ pooled,          // [32][256]
                            float* __restrict__ cnt,             // [32]
                            long long total) {
  long long gid = (long long)blockIdx.x * blockDim.x + threadIdx.x;
  if (gid >= total) return;
  long long n = gid >> 8;
  int c = (int)(gid & 255);
  float v = (float)hnode[n * 256 + c] * bn[c] + bn[256 + c];
  long long g = batch[n];
  atomicAdd(&pooled[g * 256 + c], v);
  if (c == 0) atomicAdd(&cnt[g], 1.0f);
}

// ---------------------------------------------------------------------------
// head: 32x256 -> mlp1(64, BN) -> mlp2(32, BN) -> linear(1). One block in LDS.
// ---------------------------------------------------------------------------
__global__ __launch_bounds__(256)
void head_kernel(const float* __restrict__ pooled, const float* __restrict__ cnt,
                 const float* __restrict__ W1, const float* __restrict__ b1,
                 const float* __restrict__ g1, const float* __restrict__ be1,
                 const float* __restrict__ W2, const float* __restrict__ b2,
                 const float* __restrict__ g2, const float* __restrict__ be2,
                 const float* __restrict__ oW, const float* __restrict__ ob,
                 float* __restrict__ out) {
  __shared__ float P[32 * 256];
  __shared__ float H1[32 * 64];
  __shared__ float H2[32 * 32];
  __shared__ float csc[64], csh[64];
  const int tid = threadIdx.x;

  for (int i = tid; i < 32 * 256; i += 256) {
    int g = i >> 8;
    P[i] = pooled[i] / fmaxf(cnt[g], 1.f);
  }
  __syncthreads();

  for (int i = tid; i < 32 * 64; i += 256) {
    int r = i >> 6, c = i & 63;
    float acc = b1[c];
    for (int k = 0; k < 256; k++) acc += P[r * 256 + k] * W1[k * 64 + c];
    H1[i] = fmaxf(acc, 0.f);
  }
  __syncthreads();
  if (tid < 64) {
    float s = 0.f, q = 0.f;
    for (int r = 0; r < 32; r++) { float v = H1[r * 64 + tid]; s += v; q += v * v; }
    float mean = s / 32.f, var = fmaxf(q / 32.f - mean * mean, 0.f);
    float sc = g1[tid] * rsqrtf(var + 1e-5f);
    csc[tid] = sc; csh[tid] = be1[tid] - mean * sc;
  }
  __syncthreads();
  for (int i = tid; i < 32 * 64; i += 256) { int c = i & 63; H1[i] = H1[i] * csc[c] + csh[c]; }
  __syncthreads();

  for (int i = tid; i < 32 * 32; i += 256) {
    int r = i >> 5, c = i & 31;
    float acc = b2[c];
    for (int k = 0; k < 64; k++) acc += H1[r * 64 + k] * W2[k * 32 + c];
    H2[i] = fmaxf(acc, 0.f);
  }
  __syncthreads();
  if (tid < 32) {
    float s = 0.f, q = 0.f;
    for (int r = 0; r < 32; r++) { float v = H2[r * 32 + tid]; s += v; q += v * v; }
    float mean = s / 32.f, var = fmaxf(q / 32.f - mean * mean, 0.f);
    float sc = g2[tid] * rsqrtf(var + 1e-5f);
    csc[tid] = sc; csh[tid] = be2[tid] - mean * sc;
  }
  __syncthreads();
  for (int i = tid; i < 32 * 32; i += 256) { int c = i & 31; H2[i] = H2[i] * csc[c] + csh[c]; }
  __syncthreads();
  if (tid < 32) {
    float acc = ob[0];
    for (int k = 0; k < 32; k++) acc += H2[tid * 32 + k] * oW[k];
    out[tid] = acc;
  }
}

// ---------------------------------------------------------------------------
// host launcher
// ---------------------------------------------------------------------------
extern "C" void kernel_launch(void* const* d_in, const int* in_sizes, int n_in,
                              void* d_out, int out_size, void* d_ws, size_t ws_size,
                              hipStream_t stream) {
  (void)n_in; (void)out_size; (void)ws_size;

  const float*     x     = (const float*)d_in[0];
  const long long* ei    = (const long long*)d_in[1];
  const long long* batch = (const long long*)d_in[2];
  const long long N = in_sizes[0] / 16;
  const long long E = in_sizes[1] / 2;
  const long long* srcI = ei;
  const long long* dstI = ei + E;
  auto F = [&](int i) { return (const float*)d_in[i]; };

  // ---- workspace carve (256B aligned) ----
  char* wp = (char*)d_ws;
  auto alloc = [&](size_t bytes) -> void* {
    void* r = (void*)wp;
    wp += (bytes + 255) & ~(size_t)255;
    return r;
  };
  _Float16* hedge = (_Float16*)alloc((size_t)E * 64 * sizeof(_Float16));
  unsigned* xenc  = (unsigned*)alloc((size_t)N * 64 * sizeof(unsigned));
  _Float16* x1    = (_Float16*)alloc((size_t)N * 64 * sizeof(_Float16));
  _Float16* x2    = (_Float16*)alloc((size_t)N * 64 * sizeof(_Float16));
  _Float16* x3    = (_Float16*)alloc((size_t)N * 64 * sizeof(_Float16));
  _Float16* hnode = (_Float16*)alloc((size_t)N * 256 * sizeof(_Float16));
  float*    stats = (float*)alloc(7 * 512 * sizeof(float));
  float*    bn    = (float*)alloc(7 * 512 * sizeof(float));
  const long long wsz[7]  = {32 * 64, 64 * 64, 128 * 64, 64 * 64, 128 * 64, 64 * 64, 192 * 256};
  const int       widx[7] = {3, 7, 11, 15, 19, 23, 27};
  const int       wcout[7] = {64, 64, 64, 64, 64, 64, 256};
  _Float16* wc[7];
  for (int i = 0; i < 7; i++) wc[i] = (_Float16*)alloc((size_t)wsz[i] * sizeof(_Float16));
  float* pooled = (float*)alloc((32 * 256 + 32) * sizeof(float));
  float* cnt    = pooled + 32 * 256;

  // ---- init: zero stats + pooled; swizzle weights to fragment-major f16 ----
  fill_u32<<<(7 * 512 + 255) / 256, 256, 0, stream>>>((unsigned*)stats, 0u, 7 * 512);
  fill_u32<<<(32 * 256 + 32 + 255) / 256, 256, 0, stream>>>((unsigned*)pooled, 0u, 32 * 256 + 32);
  for (int i = 0; i < 7; i++)
    weight_swizzle<<<(int)((wsz[i] + 255) / 256), 256, 0, stream>>>(F(widx[i]), wc[i],
                                                                    wcout[i], wsz[i]);

  const long long tiles = (E + 15) / 16;
  const int gE   = (int)((tiles + 7) / 8);            // edge GEMM blocks
  const int gN64 = (int)((N * 64 + 255) / 256);
  const int gN32 = (int)((N * 32 + 255) / 256);
  const int gE16 = (int)((E * 16 + 255) / 256);

  // ---- three EdgeConv layers ----
  for (int l = 0; l < 3; l++) {
    const int base = 3 + l * 8;                        // blockA params at base..base+3
    float* stA = stats + (2 * l) * 512;
    float* stB = stats + (2 * l + 1) * 512;
    float* bnA = bn + (2 * l) * 512;
    float* bnB = bn + (2 * l + 1) * 512;

    fill_u32<<<gN64, 256, 0, stream>>>(xenc, 0x007FFFFFu, N * 64);  // enc(-inf)

    if (l == 0)
      edge_gemm1<16, float><<<gE, 256, 0, stream>>>(x, srcI, dstI, wc[0], F(base + 1),
                                                    hedge, stA, E);
    else
      edge_gemm1<64, _Float16><<<gE, 256, 0, stream>>>((l == 1) ? x1 : x2, srcI, dstI,
                                                       wc[2 * l], F(base + 1), hedge, stA, E);

    bn_finalize<<<1, 256, 0, stream>>>(stA, F(base + 2), F(base + 3), (float)E, 64, bnA);
    edge_gemm2<<<gE, 256, 0, stream>>>(hedge, bnA, wc[2 * l + 1], F(base + 5), stB, E);
    bn_finalize<<<1, 256, 0, stream>>>(stB, F(base + 6), F(base + 7), (float)E, 64, bnB);
    scatter_max<<<gE16, 256, 0, stream>>>(hedge, bnB, dstI, xenc, E * 16);

    _Float16* xo = (l == 0) ? x1 : (l == 1) ? x2 : x3;
    decode_nodes<<<gN32, 256, 0, stream>>>(xenc, (unsigned*)xo, N * 32);
  }

  // ---- lin1 (192 -> 256) + BN + mean pool ----
  float* stL = stats + 6 * 512;
  float* bnL = bn + 6 * 512;
  const int gL = (int)((N + 15) / 16);
  lin1_gemm<<<gL, 128, 0, stream>>>(x1, x2, x3, wc[6], F(28), hnode, stL, N);
  bn_finalize<<<1, 256, 0, stream>>>(stL, F(29), F(30), (float)N, 256, bnL);
  pool_kernel<<<(int)((N * 256 + 255) / 256), 256, 0, stream>>>(hnode, bnL, batch,
                                                                pooled, cnt, N * 256);

  // ---- head MLP ----
  head_kernel<<<1, 256, 0, stream>>>(pooled, cnt,
                                     F(31), F(32), F(33), F(34),
                                     F(35), F(36), F(37), F(38),
                                     F(39), F(40), (float*)d_out);
}
